// AwkwardDeepSetDoubleJagged_1176821039422
// MI455X (gfx1250) — matrix operand order repaired and checked by reading, not compiled, not run on hardware
//
#include <hip/hip_runtime.h>

typedef __attribute__((ext_vector_type(16))) _Float16 v16h;
typedef __attribute__((ext_vector_type(2)))  _Float16 v2h;
typedef __attribute__((ext_vector_type(8)))  float    v8f;
typedef __attribute__((ext_vector_type(4)))  unsigned int v4u;

union Frag { v4u u[2]; v16h h; };

#define E_EVENTS 8192

__device__ __forceinline__ v8f wmma_f16(v16h a, v16h b, v8f c) {
  // D = A(16x32 f16) * B(32x16 f16) + C(16x16 f32)
  return __builtin_amdgcn_wmma_f32_16x16x32_f16(false, a, false, b, (short)0, c,
                                                false, false);
}

__device__ __forceinline__ void lds_fence() {
  asm volatile("s_wait_dscnt 0" ::: "memory");
}

// ---------------------------------------------------------------------------
// Kernel A: fused phi1 (1->64 relu, 64x64 GEMM relu) + sorted segment-sum.
// 128 threads = 4 waves. Each wave processes 16-element groups.
// ---------------------------------------------------------------------------
__global__ void __launch_bounds__(128)
phi1_segsum(const float* __restrict__ x, const int* __restrict__ seg,
            const float* __restrict__ w0, const float* __restrict__ b0,
            const float* __restrict__ w1, const float* __restrict__ b1,
            float* __restrict__ pooled, int n)
{
  __shared__ __align__(16) _Float16 w1t[64 * 64];      // W1 transposed, f16
  __shared__ __align__(16) _Float16 stage[4][16 * 64]; // per-wave A staging

  const int tid = threadIdx.x;
  // cooperative transpose W1 -> LDS (column-major so B-frag K runs are contiguous)
  for (int i = tid; i < 64 * 64; i += 128) {
    int col = i >> 6, k = i & 63;
    w1t[i] = (_Float16)w1[k * 64 + col];  // w1t[col*64 + k]
  }
  __syncthreads();

  const int lane = tid & 31;
  const int wave = tid >> 5;
  const int ln   = lane & 15;
  const int hi   = lane >> 4;

  // per-lane constants for the 1->64 layer (this lane owns cols 2*lane, 2*lane+1)
  const float w0a = w0[2 * lane], w0b = w0[2 * lane + 1];
  const float b0a = b0[2 * lane], b0b = b0[2 * lane + 1];
  float b1v[4];
#pragma unroll
  for (int nb = 0; nb < 4; ++nb) b1v[nb] = b1[nb * 16 + ln];

  // B fragments of W1, resident in registers for the whole wave.
  // B 32x16 f16 layout: lane ln holds column nb*16+ln, K (hi*8..hi*8+7, 16+hi*8..16+hi*8+7).
  Frag bf[4][2];
#pragma unroll
  for (int nb = 0; nb < 4; ++nb)
#pragma unroll
    for (int kb = 0; kb < 2; ++kb) {
      const _Float16* p = &w1t[(nb * 16 + ln) * 64 + kb * 32 + hi * 8];
      bf[nb][kb].u[0] = *(const v4u*)(p);
      bf[nb][kb].u[1] = *(const v4u*)(p + 16);
    }

  _Float16* st = &stage[wave][0];
  const int G    = n >> 4;                 // groups of 16 elements
  const int wgid = blockIdx.x * 4 + wave;
  const int nw   = gridDim.x * 4;

  for (int g = wgid; g < G; g += nw) {
    const int base = g << 4;
    const float xv = x[base + ln];  // 16 values, broadcast via shfl below

    // h1 = relu(x * w0 + b0), f16, row-major 16x64 into staging
#pragma unroll
    for (int r = 0; r < 16; ++r) {
      float xr = __shfl(xv, r, 32);
      v2h p;
      p.x = (_Float16)fmaxf(xr * w0a + b0a, 0.0f);
      p.y = (_Float16)fmaxf(xr * w0b + b0b, 0.0f);
      *(v2h*)(st + r * 64 + 2 * lane) = p;
    }
    lds_fence();

    // A fragments (16-bit A 16x32 layout: hi=0 -> K 0-7,16-23 ; hi=1 -> K 8-15,24-31)
    Frag a0, a1;
    const _Float16* rp = st + ln * 64 + hi * 8;
    a0.u[0] = *(const v4u*)(rp);
    a0.u[1] = *(const v4u*)(rp + 16);
    a1.u[0] = *(const v4u*)(rp + 32);
    a1.u[1] = *(const v4u*)(rp + 48);

    v8f cc[4] = {};
#pragma unroll
    for (int nb = 0; nb < 4; ++nb) {
      cc[nb] = wmma_f16(a0.h, bf[nb][0].h, cc[nb]);
      cc[nb] = wmma_f16(a1.h, bf[nb][1].h, cc[nb]);
    }

    // sorted segment-sum: fast path when all 16 rows share one event
    const int s0  = seg[base];
    const int s15 = seg[base + 15];
    if (s0 == s15) {
#pragma unroll
      for (int nb = 0; nb < 4; ++nb) {
        float cs = 0.0f;
#pragma unroll
        for (int r = 0; r < 8; ++r) cs += fmaxf(cc[nb][r] + b1v[nb], 0.0f);
        cs += __shfl_xor(cs, 16, 32);  // combine M 0-7 with M 8-15
        if (hi == 0) atomicAdd(&pooled[s0 * 64 + nb * 16 + ln], cs);
      }
    } else {
#pragma unroll
      for (int nb = 0; nb < 4; ++nb)
#pragma unroll
        for (int r = 0; r < 8; ++r) {
          int m = r + hi * 8;
          float v = fmaxf(cc[nb][r] + b1v[nb], 0.0f);
          atomicAdd(&pooled[seg[base + m] * 64 + nb * 16 + ln], v);
        }
    }
  }
}

// ---------------------------------------------------------------------------
// Kernel B: per-event chain rho1 -> out1 -> phi2 (5 chained 64x64 GEMMs with
// relu) fully in LDS/registers, then global sum-pool via atomics.
// 128 threads = 4 waves, one 16-event tile per wave.
// ---------------------------------------------------------------------------
__global__ void __launch_bounds__(128)
event_chain(const float* __restrict__ pooled,
            const float* __restrict__ m0, const float* __restrict__ bb0,
            const float* __restrict__ m1, const float* __restrict__ bb1,
            const float* __restrict__ m2, const float* __restrict__ bb2,
            const float* __restrict__ m3, const float* __restrict__ bb3,
            const float* __restrict__ m4, const float* __restrict__ bb4,
            float* __restrict__ sAcc)
{
  __shared__ __align__(16) _Float16 wt[5][64 * 64];   // 40KB transposed weights
  __shared__ __align__(16) _Float16 stg[4][16 * 64];  // per-wave staging

  const int tid = threadIdx.x;
  const float* mats[5] = {m0, m1, m2, m3, m4};
  for (int mi = 0; mi < 5; ++mi)
    for (int i = tid; i < 64 * 64; i += 128) {
      int col = i >> 6, k = i & 63;
      wt[mi][i] = (_Float16)mats[mi][k * 64 + col];
    }
  __syncthreads();

  const int lane = tid & 31;
  const int wave = tid >> 5;
  const int ln   = lane & 15;
  const int hi   = lane >> 4;
  const float* biases[5] = {bb0, bb1, bb2, bb3, bb4};

  const int e0 = (blockIdx.x * 4 + wave) * 16;
  _Float16* st = &stg[wave][0];

  // stage-in: pooled tile (f32 -> f16 row-major)
#pragma unroll
  for (int r = 0; r < 16; ++r) {
    v2h p;
    p.x = (_Float16)pooled[(e0 + r) * 64 + 2 * lane];
    p.y = (_Float16)pooled[(e0 + r) * 64 + 2 * lane + 1];
    *(v2h*)(st + r * 64 + 2 * lane) = p;
  }

#pragma unroll
  for (int s = 0; s < 5; ++s) {
    lds_fence();
    Frag a0, a1;
    const _Float16* rp = st + ln * 64 + hi * 8;
    a0.u[0] = *(const v4u*)(rp);
    a0.u[1] = *(const v4u*)(rp + 16);
    a1.u[0] = *(const v4u*)(rp + 32);
    a1.u[1] = *(const v4u*)(rp + 48);

    v8f nc[4] = {};
#pragma unroll
    for (int nb = 0; nb < 4; ++nb) {
#pragma unroll
      for (int kb = 0; kb < 2; ++kb) {
        Frag bfr;
        const _Float16* bp = &wt[s][(nb * 16 + ln) * 64 + kb * 32 + hi * 8];
        bfr.u[0] = *(const v4u*)(bp);
        bfr.u[1] = *(const v4u*)(bp + 16);
        nc[nb] = wmma_f16(kb ? a1.h : a0.h, bfr.h, nc[nb]);
      }
    }

    float bv[4];
#pragma unroll
    for (int nb = 0; nb < 4; ++nb) bv[nb] = biases[s][nb * 16 + ln];

    if (s < 4) {
      // relu(C + bias) back to staging for the next GEMM
#pragma unroll
      for (int nb = 0; nb < 4; ++nb)
#pragma unroll
        for (int r = 0; r < 8; ++r) {
          float v = fmaxf(nc[nb][r] + bv[nb], 0.0f);
          st[(r + hi * 8) * 64 + nb * 16 + ln] = (_Float16)v;
        }
    } else {
      // g = relu(...); sum over the 16 events of this tile -> global accumulator
#pragma unroll
      for (int nb = 0; nb < 4; ++nb) {
        float cs = 0.0f;
#pragma unroll
        for (int r = 0; r < 8; ++r) cs += fmaxf(nc[nb][r] + bv[nb], 0.0f);
        cs += __shfl_xor(cs, 16, 32);
        if (hi == 0) atomicAdd(&sAcc[nb * 16 + ln], cs);
      }
    }
  }
}

// ---------------------------------------------------------------------------
// Kernel C: tiny head  s[1,64] -> relu GEMV x2 -> [1,10]
// ---------------------------------------------------------------------------
__global__ void __launch_bounds__(64)
final_head(const float* __restrict__ sAcc,
           const float* __restrict__ w0, const float* __restrict__ b0,
           const float* __restrict__ w1, const float* __restrict__ b1,
           const float* __restrict__ ow, const float* __restrict__ ob,
           float* __restrict__ out)
{
  __shared__ float sb[64], t0[64], t1[64];
  const int t = threadIdx.x;
  sb[t] = sAcc[t];
  __syncthreads();
  float a = b0[t];
  for (int k = 0; k < 64; ++k) a += sb[k] * w0[k * 64 + t];
  t0[t] = fmaxf(a, 0.0f);
  __syncthreads();
  a = b1[t];
  for (int k = 0; k < 64; ++k) a += t0[k] * w1[k * 64 + t];
  t1[t] = fmaxf(a, 0.0f);
  __syncthreads();
  if (t < 10) {
    float o = ob[t];
    for (int k = 0; k < 64; ++k) o += t1[k] * ow[k * 10 + t];
    out[t] = o;
  }
}

__global__ void zero_kernel(float* __restrict__ p, int n) {
  int i = blockIdx.x * blockDim.x + threadIdx.x;
  if (i < n) p[i] = 0.0f;
}

// ---------------------------------------------------------------------------
extern "C" void kernel_launch(void* const* d_in, const int* in_sizes, int n_in,
                              void* d_out, int out_size, void* d_ws, size_t ws_size,
                              hipStream_t stream)
{
  const float* x    = (const float*)d_in[0];
  const int*   seg  = (const int*)d_in[1];
  const float* p1w0 = (const float*)d_in[2];
  const float* p1b0 = (const float*)d_in[3];
  const float* p1w1 = (const float*)d_in[4];
  const float* p1b1 = (const float*)d_in[5];
  const float* r1w0 = (const float*)d_in[6];
  const float* r1b0 = (const float*)d_in[7];
  const float* r1w1 = (const float*)d_in[8];
  const float* r1b1 = (const float*)d_in[9];
  const float* o1w  = (const float*)d_in[10];
  const float* o1b  = (const float*)d_in[11];
  const float* p2w0 = (const float*)d_in[12];
  const float* p2b0 = (const float*)d_in[13];
  const float* p2w1 = (const float*)d_in[14];
  const float* p2b1 = (const float*)d_in[15];
  const float* r2w0 = (const float*)d_in[16];
  const float* r2b0 = (const float*)d_in[17];
  const float* r2w1 = (const float*)d_in[18];
  const float* r2b1 = (const float*)d_in[19];
  const float* o2w  = (const float*)d_in[20];
  const float* o2b  = (const float*)d_in[21];

  const int n = in_sizes[0];
  float* pooled = (float*)d_ws;                      // [E, 64] f32
  float* sAcc   = pooled + (size_t)E_EVENTS * 64;    // [64] f32
  float* out    = (float*)d_out;                     // [10] f32

  const int zn = E_EVENTS * 64 + 64;
  zero_kernel<<<(zn + 255) / 256, 256, 0, stream>>>(pooled, zn);
  phi1_segsum<<<1024, 128, 0, stream>>>(x, seg, p1w0, p1b0, p1w1, p1b1, pooled, n);
  event_chain<<<E_EVENTS / 64, 128, 0, stream>>>(pooled,
                                                 r1w0, r1b0, r1w1, r1b1,
                                                 o1w, o1b, p2w0, p2b0, p2w1, p2b1,
                                                 sAcc);
  final_head<<<1, 64, 0, stream>>>(sAcc, r2w0, r2b0, r2w1, r2b1, o2w, o2b, out);
}
